// FP8Linear_74680891343685
// MI455X (gfx1250) — matrix operand order repaired and checked by reading, not compiled, not run on hardware
//
#include <hip/hip_runtime.h>

// ---------------------------------------------------------------------------
// FP8-valued Linear with FWHT front-end, tuned for MI455X (gfx1250, wave32).
//   out = FWHT(x) @ W^T + b     T=8192, D=2048, O=8192
// Math path: f16 WMMA (v_wmma_f32_16x16x32_f16) with fp32 accumulation.
// Weights are exactly representable in f16 (they were FP8-E4M3 quantized).
//
// GEMM tiling (v2): block = 128 M x 256 N, 8 waves (2 M x 4 N),
// wave tile = 64x64 = 4x4 fragments -> 16 WMMA per 32-wide K step,
// 8 LDS fragment loads per step (16 B/lane/WMMA, 1.5x better than v1).
// ---------------------------------------------------------------------------

#define TOKENS 8192
#define D_IN   2048
#define D_OUT  8192

typedef __attribute__((ext_vector_type(16))) _Float16 v16h;
typedef __attribute__((ext_vector_type(8)))  _Float16 v8h;
typedef __attribute__((ext_vector_type(4)))  _Float16 v4h;
typedef __attribute__((ext_vector_type(8)))  float    v8f;

union V16U { v16h v; v8h h[2]; };

// ---------------------------------------------------------------------------
// Kernel 1: FWHT of one 2048-float row in LDS, scale by 1/sqrt(2048),
// write out as f16 (row-major, K contiguous).
// ---------------------------------------------------------------------------
__global__ void __launch_bounds__(256)
fwht_f16_kernel(const float* __restrict__ x, _Float16* __restrict__ xh) {
    __shared__ float row[D_IN];
    const int tid = threadIdx.x;
    const size_t base = (size_t)blockIdx.x * D_IN;

    #pragma unroll
    for (int i = 0; i < D_IN / 256; ++i)
        row[tid + i * 256] = x[base + tid + i * 256];

    // 11 butterfly stages (matches reference: [a+b, a-b] with half-stride h)
    for (int h = 1; h < D_IN; h <<= 1) {
        __syncthreads();
        #pragma unroll
        for (int i = 0; i < (D_IN / 2) / 256; ++i) {
            const int p   = tid + i * 256;        // 0 .. 1023
            const int grp = p / h;
            const int idx = p - grp * h;
            const int a   = grp * 2 * h + idx;
            const int b   = a + h;
            const float va = row[a];
            const float vb = row[b];
            row[a] = va + vb;
            row[b] = va - vb;
        }
    }
    __syncthreads();

    const float scale = 0.022097086912079612f;    // 1/sqrt(2048)
    #pragma unroll
    for (int i = 0; i < D_IN / 256; ++i) {
        const int j = tid + i * 256;
        xh[base + j] = (_Float16)(row[j] * scale);
    }
}

// ---------------------------------------------------------------------------
// Kernel 2: weight fp32 -> f16 (exact: values are FP8-E4M3 quantized).
// ---------------------------------------------------------------------------
__global__ void __launch_bounds__(256)
cvt_w_kernel(const float4* __restrict__ w, _Float16* __restrict__ wh) {
    const size_t i = (size_t)blockIdx.x * blockDim.x + threadIdx.x;
    const float4 v = w[i];
    v4h o;
    o[0] = (_Float16)v.x; o[1] = (_Float16)v.y;
    o[2] = (_Float16)v.z; o[3] = (_Float16)v.w;
    *(v4h*)(wh + i * 4) = o;
}

// ---------------------------------------------------------------------------
// Kernel 3: WMMA GEMM  out[t,o] = sum_d xh[t,d]*wh[o,d] + bias[o]
// Block: 256 threads = 8 waves, 128 M x 256 N output tile.
// Waves 2 (M) x 4 (N); each wave owns 64x64 = 4x4 16x16 fragments.
// Per 32-wide K step: stage X tile (128x32 halves) + W tile (256x32 halves)
// into LDS; each wave does 8 fragment loads and 16 WMMAs.
// NT layout => A frag = two contiguous 8-half chunks / lane,
//              B frag = two contiguous 8-half chunks / lane. No transpose.
// ---------------------------------------------------------------------------
__global__ void __launch_bounds__(256)
gemm_wmma_kernel(const _Float16* __restrict__ xh,
                 const _Float16* __restrict__ wh,
                 const float*    __restrict__ bias,
                 float*          __restrict__ out) {
    __shared__ _Float16 lx[128 * 32];   // X tile: 128 rows x 32 K halves (8 KB)
    __shared__ _Float16 lw[256 * 32];   // W tile: 256 rows x 32 K halves (16 KB)

    const int tid   = threadIdx.x;
    const int lane  = tid & 31;
    const int wv    = tid >> 5;
    const int m_off = (wv & 1) * 64;    // wave M offset within 128
    const int n_off = (wv >> 1) * 64;   // wave N offset within 256
    const int t_base = blockIdx.y * 128;
    const int o_base = blockIdx.x * 256;

    const int rsel = lane >> 4;         // 0: lanes 0-15, 1: lanes 16-31
    const int rlan = lane & 15;

    // staging: X -> each thread 16 halves (2 x b128); W -> one full row (4 x b128)
    const int xrow = tid >> 1;          // 0..127
    const int xcol = (tid & 1) * 16;    // 0 or 16

    v8f acc[4][4] = {};

    for (int kb = 0; kb < D_IN; kb += 32) {
        __syncthreads();   // protect LDS tiles from previous iteration's reads

        const uint4* gx = (const uint4*)(xh + (size_t)(t_base + xrow) * D_IN + kb + xcol);
        const uint4* gw = (const uint4*)(wh + (size_t)(o_base + tid) * D_IN + kb);
        uint4* sx = (uint4*)(lx + xrow * 32 + xcol);
        uint4* sw = (uint4*)(lw + tid * 32);
        sx[0] = gx[0]; sx[1] = gx[1];
        sw[0] = gw[0]; sw[1] = gw[1]; sw[2] = gw[2]; sw[3] = gw[3];

        __syncthreads();

        // A fragments (16x32, MxK): lane holds row M = lane%16;
        // halves 0..7 at K = 8*rsel, halves 8..15 at K = 16 + 8*rsel.
        V16U a[4];
        #pragma unroll
        for (int mi = 0; mi < 4; ++mi) {
            const _Float16* p = lx + (m_off + mi * 16 + rlan) * 32 + rsel * 8;
            a[mi].h[0] = *(const v8h*)(p);
            a[mi].h[1] = *(const v8h*)(p + 16);
        }
        // B fragments (32x16, KxN): lane holds col N = lane%16;
        // halves 0..7 at K = 8*rsel, halves 8..15 at K = 16 + 8*rsel.
        V16U bfr[4];
        #pragma unroll
        for (int ni = 0; ni < 4; ++ni) {
            const _Float16* p = lw + (n_off + ni * 16 + rlan) * 32 + rsel * 8;
            bfr[ni].h[0] = *(const v8h*)(p);
            bfr[ni].h[1] = *(const v8h*)(p + 16);
        }

        #pragma unroll
        for (int mi = 0; mi < 4; ++mi)
            #pragma unroll
            for (int ni = 0; ni < 4; ++ni)
                acc[mi][ni] = __builtin_amdgcn_wmma_f32_16x16x32_f16(
                    /*neg_a=*/false, a[mi].v,
                    /*neg_b=*/false, bfr[ni].v,
                    /*c_mod=*/(short)0, acc[mi][ni],
                    /*reuse_a=*/false, /*reuse_b=*/false);
    }

    // Epilogue: C/D layout is VGPR r -> M = r + 8*rsel, N = rlan. Add bias.
    #pragma unroll
    for (int ni = 0; ni < 4; ++ni) {
        const int col = o_base + n_off + ni * 16 + rlan;
        const float bv = bias[col];
        #pragma unroll
        for (int mi = 0; mi < 4; ++mi) {
            const int row0 = t_base + m_off + mi * 16 + rsel * 8;
            #pragma unroll
            for (int r = 0; r < 8; ++r)
                out[(size_t)(row0 + r) * D_OUT + col] = acc[mi][ni][r] + bv;
        }
    }
}

// ---------------------------------------------------------------------------
extern "C" void kernel_launch(void* const* d_in, const int* in_sizes, int n_in,
                              void* d_out, int out_size, void* d_ws, size_t ws_size,
                              hipStream_t stream) {
    const float* x    = (const float*)d_in[0];
    const float* w    = (const float*)d_in[1];
    const float* bias = (const float*)d_in[2];
    float* out = (float*)d_out;

    // workspace: xh_f16 (32 MB) | wh_f16 (32 MB)
    _Float16* xh = (_Float16*)d_ws;
    _Float16* wh = xh + (size_t)TOKENS * D_IN;

    // 1) FWHT + cast to f16
    fwht_f16_kernel<<<TOKENS, 256, 0, stream>>>(x, xh);

    // 2) weight fp32 -> f16 (exact)
    const int cvt_blocks = (D_OUT * D_IN) / (256 * 4);   // 16384
    cvt_w_kernel<<<cvt_blocks, 256, 0, stream>>>((const float4*)w, wh);

    // 3) WMMA GEMM + bias
    dim3 grid(D_OUT / 256, TOKENS / 128);                // 32 x 64
    gemm_wmma_kernel<<<grid, 256, 0, stream>>>(xh, wh, bias, out);
}
